// GaussRegisterStep_11373073400447
// MI455X (gfx1250) — matrix operand order repaired and checked by reading, not compiled, not run on hardware
//
#include <hip/hip_runtime.h>
#include <hip/hip_bf16.h>
#include <math.h>

typedef __bf16 bf16;
typedef __attribute__((ext_vector_type(16))) __bf16 v16bf;
typedef __attribute__((ext_vector_type(8)))  __bf16 v8bf;
typedef __attribute__((ext_vector_type(8)))  float   v8f;

#define B_  4
#define T_  2048
#define V_  2048
#define NF_ 512
#define NC_ 1024
#define M_TOT (B_ * T_)          // 8192
#define EPS_ 1.1920929e-07f

// ---------------------------------------------------------------------------
// CDNA5 async DMA: global -> LDS, 16B per lane, tracked by ASYNCcnt.
// lds argument is the 32-bit wave-relative LDS byte offset (low 32 bits of
// the generic pointer to a __shared__ object).
// ---------------------------------------------------------------------------
__device__ __forceinline__ void async_ld_b128(bf16* lds_p, const bf16* g) {
    unsigned lds_off = (unsigned)(size_t)lds_p;
    asm volatile("global_load_async_to_lds_b128 %0, %1, off"
                 :: "v"(lds_off), "v"(g) : "memory");
}
__device__ __forceinline__ void wait_async0() {
    asm volatile("s_wait_asynccnt 0x0" ::: "memory");
}

// ---------------------------------------------------------------------------
// DFT basis generation: Pt[c][v] (analysis, [1024][2048]) and St[v][c]
// (synthesis, [2048][1024]). Exact integer phase reduction mod 2048.
// ---------------------------------------------------------------------------
__global__ void k_basis(bf16* __restrict__ Pt, bf16* __restrict__ St) {
    long idx = (long)blockIdx.x * 256 + threadIdx.x;   // 1024*2048 total
    int c = (int)(idx >> 11);          // 0..1023 (channel)
    int v = (int)(idx & 2047);         // 0..2047 (vocab pos)
    int f = (c < 512) ? (c + 1) : (c - 511);           // freq bin 1..512
    int ph = (int)(((long)f * (long)v) & 2047);        // exact mod V
    float ang = (float)ph * (6.283185307179586f / 2048.f);
    float val = (c < 512) ? cosf(ang) : -sinf(ang);
    Pt[((long)c << 11) | v] = (bf16)val;
    St[((long)v << 10) | c] = (bf16)(val * (2.0f / 2048.0f));
}

// f32 -> bf16 straight convert
__global__ void k_cvt(const float* __restrict__ in, bf16* __restrict__ out, int n) {
    int i = blockIdx.x * 256 + threadIdx.x;
    if (i < n) out[i] = (bf16)in[i];
}

// f32 [n][n] -> bf16 transposed [n][n]
__global__ void k_tr(const float* __restrict__ in, bf16* __restrict__ out, int n) {
    __shared__ float tile[32][33];
    int bx = blockIdx.x * 32, by = blockIdx.y * 32;
    int tx = threadIdx.x, ty = threadIdx.y;            // 32 x 8
    #pragma unroll
    for (int i = 0; i < 32; i += 8)
        tile[ty + i][tx] = in[(long)(by + ty + i) * n + bx + tx];
    __syncthreads();
    #pragma unroll
    for (int i = 0; i < 32; i += 8)
        out[(long)(bx + ty + i) * n + by + tx] = (bf16)tile[tx][ty + i];
}

// RMS norm over last dim (2048), f32 in -> bf16 out, one row per block
__global__ __launch_bounds__(256) void k_rms(const float* __restrict__ in,
                                             bf16* __restrict__ out) {
    __shared__ float red[256];
    const long row = blockIdx.x;
    const float* r = in + row * 2048;
    float v[8]; float s = 0.f;
    #pragma unroll
    for (int i = 0; i < 8; ++i) { v[i] = r[threadIdx.x + i * 256]; s += v[i] * v[i]; }
    red[threadIdx.x] = s; __syncthreads();
    for (int st = 128; st > 0; st >>= 1) {
        if (threadIdx.x < st) red[threadIdx.x] += red[threadIdx.x + st];
        __syncthreads();
    }
    float scale = rsqrtf(red[0] * (1.0f / 2048.0f) + EPS_);
    bf16* o = out + row * 2048;
    #pragma unroll
    for (int i = 0; i < 8; ++i) o[threadIdx.x + i * 256] = (bf16)(v[i] * scale);
}

// ---------------------------------------------------------------------------
// WMMA GEMM: C[M,N](f32 acc) = A[M,K](bf16,row-major) x Bt[N,K](bf16,row-major)^T
// Block tile 128x128x32, 256 threads = 8 waves (2x4), wave tile 64x32.
// LDS double-buffered, filled by GLOBAL_LOAD_ASYNC_TO_LDS_B128, one barrier
// per K-step; next tile's DMA overlaps current tile's WMMAs.
// Epilogues:
//   0: store bf16 C[m][n]
//   1: store bf16 transposed C[n][m]  (row stride = M)
//   2: causal decay weight (p0 = decay_logit), store bf16
//   3: f32 store C = resid + (p0[0]*p1[0]) * acc
//   4: bias + exact GELU, store bf16
// ---------------------------------------------------------------------------
#define BM 128
#define BN 128
#define BK 32
#define LSH 40   // padded LDS row stride in bf16 elements (80B)

template<int EPI>
__global__ __launch_bounds__(256)
void k_gemm(const bf16* __restrict__ A, const bf16* __restrict__ Bt,
            void* __restrict__ Cv, int M, int N, int K,
            const float* __restrict__ resid,
            const float* __restrict__ p0, const float* __restrict__ p1,
            const float* __restrict__ bias) {
    __shared__ bf16 As[2][BM * LSH];
    __shared__ bf16 Bs[2][BN * LSH];

    const int tid  = threadIdx.x;
    const int lane = tid & 31;
    const int w    = tid >> 5;
    const int wrow = w >> 2;          // 0..1  -> 64-row wave tile
    const int wcol = w & 3;           // 0..3  -> 32-col wave tile
    const int m0 = blockIdx.y * BM;
    const int n0 = blockIdx.x * BN;

    const int ldRow   = tid >> 2;           // 0..63
    const int ldChunk = (tid & 3) * 8;      // bf16 element offset within 32

    v8f acc[4][2];
    #pragma unroll
    for (int i = 0; i < 4; ++i)
        #pragma unroll
        for (int j = 0; j < 2; ++j)
            acc[i][j] = (v8f)(0.0f);

    const int am   = lane & 15;
    const int kh8  = (lane >> 4) * 8;    // A frag K base (elems)
    const int kh16 = (lane >> 4) * 8;    // B frag K base (elems)

    // per-thread staging source rows (fixed), two 64-row halves each
    const bf16* gA0 = A  + (long)(m0 + ldRow) * K + ldChunk;
    const bf16* gA1 = A  + (long)(m0 + ldRow + 64) * K + ldChunk;
    const bf16* gB0 = Bt + (long)(n0 + ldRow) * K + ldChunk;
    const bf16* gB1 = Bt + (long)(n0 + ldRow + 64) * K + ldChunk;
    const int ldsOff0 = ldRow * LSH + ldChunk;
    const int ldsOff1 = (ldRow + 64) * LSH + ldChunk;

    // prologue: DMA tile 0 into buffer 0
    async_ld_b128(&As[0][ldsOff0], gA0);
    async_ld_b128(&Bs[0][ldsOff0], gB0);
    async_ld_b128(&As[0][ldsOff1], gA1);
    async_ld_b128(&Bs[0][ldsOff1], gB1);

    for (int kt = 0; kt < K; kt += BK) {
        const int cur = (kt >> 5) & 1;
        wait_async0();          // this wave's DMA for tile kt has landed
        __syncthreads();        // all waves' portions visible; prev reads done

        if (kt + BK < K) {      // overlap next tile's DMA with this compute
            const int nxt = cur ^ 1;
            async_ld_b128(&As[nxt][ldsOff0], gA0 + kt + BK);
            async_ld_b128(&Bs[nxt][ldsOff0], gB0 + kt + BK);
            async_ld_b128(&As[nxt][ldsOff1], gA1 + kt + BK);
            async_ld_b128(&Bs[nxt][ldsOff1], gB1 + kt + BK);
        }

        // ---- fragments (ISA 7.12.2 16-bit A 16x32 / B[n][k] layouts) ----
        v16bf af[4], bfr[2];
        #pragma unroll
        for (int i = 0; i < 4; ++i) {
            const bf16* p = &As[cur][(wrow * 64 + i * 16 + am) * LSH + kh8];
            union { v16bf v; struct { v8bf lo, hi; } s; } u;
            u.s.lo = *(const v8bf*)p;          // K = kh8 .. kh8+7
            u.s.hi = *(const v8bf*)(p + 16);   // K = kh8+16 .. kh8+23
            af[i] = u.v;
        }
        #pragma unroll
        for (int j = 0; j < 2; ++j) {
            const bf16* p = &Bs[cur][(wcol * 32 + j * 16 + am) * LSH + kh16];
            union { v16bf v; struct { v8bf lo, hi; } s; } u;
            u.s.lo = *(const v8bf*)p;          // K = kh16 .. kh16+7
            u.s.hi = *(const v8bf*)(p + 16);   // K = kh16+16 .. kh16+23
            bfr[j] = u.v;
        }

        #pragma unroll
        for (int i = 0; i < 4; ++i)
            #pragma unroll
            for (int j = 0; j < 2; ++j)
                acc[i][j] = __builtin_amdgcn_wmma_f32_16x16x32_bf16(
                    false, af[i], false, bfr[j], (short)0, acc[i][j], false, false);
    }

    // ---- epilogue ----
    const int mlh = (lane >> 4) * 8;
    const int nl  = lane & 15;
    float e0 = 0.f;
    if constexpr (EPI == 2) {
        float d = 1.0f / (1.0f + expf(-p0[0]));
        e0 = log2f(d);
    } else if constexpr (EPI == 3) {
        e0 = p0[0] * p1[0];
    }

    #pragma unroll
    for (int i = 0; i < 4; ++i) {
        #pragma unroll
        for (int j = 0; j < 2; ++j) {
            const int mb = m0 + wrow * 64 + i * 16 + mlh;
            const int nn = n0 + wcol * 32 + j * 16 + nl;
            #pragma unroll
            for (int r = 0; r < 8; ++r) {
                const int m = mb + r;
                float val = acc[i][j][r];
                if constexpr (EPI == 0) {
                    ((bf16*)Cv)[(long)m * N + nn] = (bf16)val;
                } else if constexpr (EPI == 1) {
                    ((bf16*)Cv)[(long)nn * M + m] = (bf16)val;
                } else if constexpr (EPI == 2) {
                    int diff = nn - m;  // s - t
                    float wgt = (diff > 0) ? exp2f((float)(diff - 1) * e0) : 0.0f;
                    ((bf16*)Cv)[(long)m * N + nn] = (bf16)(val * wgt);
                } else if constexpr (EPI == 3) {
                    ((float*)Cv)[(long)m * N + nn] =
                        resid[(long)m * N + nn] + e0 * val;
                } else {
                    float h = val + bias[nn];
                    float g = 0.5f * h * (1.0f + erff(h * 0.7071067811865475f));
                    ((bf16*)Cv)[(long)m * N + nn] = (bf16)g;
                }
            }
        }
    }
}

// ---------------------------------------------------------------------------
extern "C" void kernel_launch(void* const* d_in, const int* in_sizes, int n_in,
                              void* d_out, int out_size, void* d_ws, size_t ws_size,
                              hipStream_t stream) {
    const float* x      = (const float*)d_in[0];
    const float* qw     = (const float*)d_in[1];
    const float* kw     = (const float*)d_in[2];
    const float* vw     = (const float*)d_in[3];
    const float* ow     = (const float*)d_in[4];
    const float* dlog   = (const float*)d_in[5];
    const float* m_os   = (const float*)d_in[6];
    const float* f2c    = (const float*)d_in[7];
    const float* cmix   = (const float*)d_in[8];
    const float* bias   = (const float*)d_in[9];
    const float* c2f    = (const float*)d_in[10];
    const float* o_os   = (const float*)d_in[11];
    const float* m_sc   = (const float*)d_in[12];
    const float* o_sc   = (const float*)d_in[13];
    float* out = (float*)d_out;

    char* ws = (char*)d_ws;
    size_t off = 0;
    auto alloc = [&](size_t bytes) -> void* {
        void* p = ws + off;
        off = (off + bytes + 255) & ~(size_t)255;
        return p;
    };
    bf16* Pt   = (bf16*)alloc((size_t)NC_ * V_ * 2);       // [1024][2048]
    bf16* St   = (bf16*)alloc((size_t)V_ * NC_ * 2);       // [2048][1024]
    bf16* qw_b = (bf16*)alloc((size_t)NC_ * NC_ * 2);
    bf16* kw_b = (bf16*)alloc((size_t)NC_ * NC_ * 2);
    bf16* vw_b = (bf16*)alloc((size_t)NC_ * NC_ * 2);
    bf16* f2cb = (bf16*)alloc((size_t)NC_ * NC_ * 2);
    bf16* c2fb = (bf16*)alloc((size_t)NC_ * NC_ * 2);
    bf16* ow_t = (bf16*)alloc((size_t)NC_ * NC_ * 2);
    bf16* cm_t = (bf16*)alloc((size_t)NC_ * NC_ * 2);
    bf16* xn   = (bf16*)alloc((size_t)M_TOT * V_ * 2);     // rms-normed x
    bf16* Xri  = (bf16*)alloc((size_t)M_TOT * NC_ * 2);
    bf16* qb   = (bf16*)alloc((size_t)M_TOT * NC_ * 2);
    bf16* kb   = (bf16*)alloc((size_t)M_TOT * NC_ * 2);
    bf16* vt   = (bf16*)alloc((size_t)M_TOT * NC_ * 2);    // per-batch [1024][2048]
    bf16* sc   = (bf16*)alloc((size_t)T_ * T_ * 2);        // one batch of scores
    bf16* retr = (bf16*)alloc((size_t)M_TOT * NC_ * 2);
    (void)ws_size; (void)n_in; (void)in_sizes; (void)out_size;

    const int NMAT = NC_ * NC_;
    dim3 blk256(256);

    // 0) bases + weight conversions
    k_basis<<<(NC_ * V_) / 256, blk256, 0, stream>>>(Pt, St);
    k_cvt<<<(NMAT + 255) / 256, blk256, 0, stream>>>(qw, qw_b, NMAT);
    k_cvt<<<(NMAT + 255) / 256, blk256, 0, stream>>>(kw, kw_b, NMAT);
    k_cvt<<<(NMAT + 255) / 256, blk256, 0, stream>>>(vw, vw_b, NMAT);
    k_cvt<<<(NMAT + 255) / 256, blk256, 0, stream>>>(f2c, f2cb, NMAT);
    k_cvt<<<(NMAT + 255) / 256, blk256, 0, stream>>>(c2f, c2fb, NMAT);
    dim3 trg(NC_ / 32, NC_ / 32), trb(32, 8);
    k_tr<<<trg, trb, 0, stream>>>(ow, ow_t, NC_);
    k_tr<<<trg, trb, 0, stream>>>(cmix, cm_t, NC_);

    auto gemm_grid = [](int M, int N) { return dim3(N / BN, M / BM, 1); };

    // ---- Stage 1: memory block ----
    k_rms<<<M_TOT, blk256, 0, stream>>>(x, xn);
    // X_ri = xn @ P
    k_gemm<0><<<gemm_grid(M_TOT, NC_), blk256, 0, stream>>>(
        xn, Pt, Xri, M_TOT, NC_, V_, nullptr, nullptr, nullptr, nullptr);
    // q, k
    k_gemm<0><<<gemm_grid(M_TOT, NC_), blk256, 0, stream>>>(
        Xri, qw_b, qb, M_TOT, NC_, NC_, nullptr, nullptr, nullptr, nullptr);
    k_gemm<0><<<gemm_grid(M_TOT, NC_), blk256, 0, stream>>>(
        Xri, kw_b, kb, M_TOT, NC_, NC_, nullptr, nullptr, nullptr, nullptr);
    // v, stored transposed per batch: vt_b = [1024][2048]
    for (int b = 0; b < B_; ++b) {
        k_gemm<1><<<gemm_grid(T_, NC_), blk256, 0, stream>>>(
            Xri + (long)b * T_ * NC_, vw_b, vt + (long)b * NC_ * T_,
            T_, NC_, NC_, nullptr, nullptr, nullptr, nullptr);
    }
    // attention per batch: scores (decay-weighted) then retrieved
    for (int b = 0; b < B_; ++b) {
        k_gemm<2><<<gemm_grid(T_, T_), blk256, 0, stream>>>(
            qb + (long)b * T_ * NC_, kb + (long)b * T_ * NC_, sc,
            T_, T_, NC_, nullptr, dlog, nullptr, nullptr);
        k_gemm<0><<<gemm_grid(T_, NC_), blk256, 0, stream>>>(
            sc, vt + (long)b * NC_ * T_, retr + (long)b * T_ * NC_,
            T_, NC_, T_, nullptr, nullptr, nullptr, nullptr);
    }
    // Y_ri = retr @ ow   (reuse kb)
    k_gemm<0><<<gemm_grid(M_TOT, NC_), blk256, 0, stream>>>(
        retr, ow_t, kb, M_TOT, NC_, NC_, nullptr, nullptr, nullptr, nullptr);
    // x1 = x + (mem_scale*mem_out_scale) * (Y_ri @ S)  -> d_out (f32)
    k_gemm<3><<<gemm_grid(M_TOT, V_), blk256, 0, stream>>>(
        kb, St, out, M_TOT, V_, NC_, x, m_sc, m_os, nullptr);

    // ---- Stage 2: register op ----
    k_rms<<<M_TOT, blk256, 0, stream>>>(out, xn);
    k_gemm<0><<<gemm_grid(M_TOT, NC_), blk256, 0, stream>>>(
        xn, Pt, Xri, M_TOT, NC_, V_, nullptr, nullptr, nullptr, nullptr);
    // h = X_ri @ f2c^T  (reuse qb)
    k_gemm<0><<<gemm_grid(M_TOT, NC_), blk256, 0, stream>>>(
        Xri, f2cb, qb, M_TOT, NC_, NC_, nullptr, nullptr, nullptr, nullptr);
    // g = gelu(h @ channel_mix + bias)  (reuse kb)
    k_gemm<4><<<gemm_grid(M_TOT, NC_), blk256, 0, stream>>>(
        qb, cm_t, kb, M_TOT, NC_, NC_, nullptr, nullptr, nullptr, bias);
    // Y_ri2 = g @ c2f^T  (reuse qb)
    k_gemm<0><<<gemm_grid(M_TOT, NC_), blk256, 0, stream>>>(
        kb, c2fb, qb, M_TOT, NC_, NC_, nullptr, nullptr, nullptr, nullptr);
    // out = out + (op_scale*op_out_scale) * (Y_ri2 @ S)
    k_gemm<3><<<gemm_grid(M_TOT, V_), blk256, 0, stream>>>(
        qb, St, out, M_TOT, V_, NC_, out, o_sc, o_os, nullptr);
}
